// PlaneDynamicDispFields_3298534884025
// MI455X (gfx1250) — matrix operand order repaired and checked by reading, not compiled, not run on hardware
//
#include <hip/hip_runtime.h>
#include <hip/hip_bf16.h>

typedef __attribute__((ext_vector_type(16))) _Float16 v16h;
typedef __attribute__((ext_vector_type(8)))  _Float16 v8h;
typedef __attribute__((ext_vector_type(2)))  _Float16 v2h;
typedef __attribute__((ext_vector_type(8)))  float    v8f;
typedef __attribute__((ext_vector_type(4)))  int      v4i;

typedef v4i __attribute__((address_space(1)))* gas_v4i_p;  // global ("__device__") int4*
typedef v4i __attribute__((address_space(3)))* las_v4i_p;  // LDS int4*

#define NPTS        1048576
#define RESO        512
#define FEAT        64
#define PLANE_HALFS (512u * 512u * 64u)        /* 16777216 halfs = 32 MB */
#define WS_DEFORM_H 0u
#define WS_CANON_H  16777216u
#define WS_WEIGHTS_H 33554432u                 /* byte offset 64 MB */

#define W_HALFS      47104                     /* swizzled weight blob, halfs   */
#define W_CHUNKS     5888                      /* 16-byte chunks (=256*23)      */
#define LDS_W_BYTES  94208
#define LDS_ACT_BYTES 51200                    /* 128 rows * 200 halfs * 2B     */
#define LDS_TOTAL    (LDS_W_BYTES + LDS_ACT_BYTES + 1024)

#define AABB_MIN_F (-1.1f)
#define AABB_MAX_F (1.1f)
#define PI_F 3.14159265358979f

#if defined(__has_builtin)
#if __has_builtin(__builtin_amdgcn_global_load_async_to_lds_b128)
#define USE_ASYNC_LDS 1
#endif
#endif
#ifndef USE_ASYNC_LDS
#define USE_ASYNC_LDS 0
#endif

// ---- layer tables (8 matmuls: dec 0..5, can 6..7) ----
__constant__ int cK[8]   = {104, 64, 168, 64, 168, 64, 64, 64};
__constant__ int cN[8]   = {64, 64, 64, 64, 64, 2, 64, 3};
__constant__ int cNKT[8] = {4, 2, 6, 2, 6, 2, 2, 2};
__constant__ int cNNT[8] = {4, 4, 4, 4, 4, 1, 4, 1};
__constant__ unsigned cWOFF[8] = {0, 8192, 12288, 24576, 28672, 40960, 41984, 46080};

struct WPtrs {
  const float* W[8];
  const float* B[8];
};

// ------------------------------------------------------------------
// Pre-pass 1: CHW f32 planes -> HWC f16 in workspace (LDS-tiled transpose)
// ------------------------------------------------------------------
__global__ __launch_bounds__(256)
void plane_to_hwc_f16(const float* __restrict__ defp,
                      const float* __restrict__ canp,
                      _Float16* __restrict__ out) {
  __shared__ float tile[32][33];
  int b  = blockIdx.x;
  int xt = b & 15;            // 16 x-tiles of 32
  int ct = (b >> 4) & 1;      // 2 c-tiles of 32
  int y  = (b >> 5) & 511;
  int p  = (b >> 14) & 1;
  const float* src = p ? canp : defp;
  _Float16* dst = out + (size_t)p * PLANE_HALFS;
  int tx = threadIdx.x & 31;
  int tr = threadIdx.x >> 5;  // 0..7
#pragma unroll
  for (int i = 0; i < 4; i++) {
    int c = tr + i * 8;       // local channel 0..31
    tile[c][tx] = src[(size_t)(ct * 32 + c) * 262144u + (size_t)y * 512 + (xt * 32 + tx)];
  }
  __syncthreads();
#pragma unroll
  for (int i = 0; i < 4; i++) {
    int xl = tr + i * 8;      // local x 0..31
    dst[((size_t)y * 512 + xt * 32 + xl) * 64 + ct * 32 + tx] = (_Float16)tile[tx][xl];
  }
}

// ------------------------------------------------------------------
// Pre-pass 2: swizzle all weights (f32 row-major [K,N]) into the WMMA
// 32x16 B-tile lane layout, zero-padded, stored as f16 in workspace.
// tile element e: l = e>>4 (lane), j = e&15
//   k = kt*32 + (l>=16 ? 8:0) + (j<8 ? j : j+8) ; n = nt*16 + (l&15)
// ------------------------------------------------------------------
__global__ __launch_bounds__(512)
void swizzle_weights(WPtrs P, _Float16* __restrict__ wout) {
  int t = blockIdx.x;
  int layer = 7, base = 0;
  for (int i = 0; i < 8; i++) {
    int cnt = cNKT[i] * cNNT[i];
    if (t < base + cnt) { layer = i; break; }
    base += cnt;
  }
  int lt = t - base;
  int nnt = cNNT[layer];
  int nt = lt % nnt;
  int kt = lt / nnt;
  int e = threadIdx.x;
  int l = e >> 4;
  int j = e & 15;
  int k = kt * 32 + ((l >= 16) ? 8 : 0) + (j < 8 ? j : j + 8);
  int n = nt * 16 + (l & 15);
  float v = 0.f;
  if (k < cK[layer] && n < cN[layer]) v = P.W[layer][(size_t)k * cN[layer] + n];
  wout[cWOFF[layer] + (size_t)(kt * nnt + nt) * 512 + e] = (_Float16)v;
}

// ------------------------------------------------------------------
// Per-wave matmul: A = 16 activation rows (f16, LDS), B = pre-swizzled
// weight tiles staged in LDS, C/D = f32 accumulators. Fully unrolled.
// ------------------------------------------------------------------
template <int NKT_, int NTI_>
__device__ __forceinline__ void wave_mm(const _Float16 (*act)[200],
                                        const _Float16* wTiles,
                                        int kbase, int wv, int lane,
                                        v8f (&acc)[NTI_]) {
  const int row  = wv * 16 + (lane & 15);
  const int koff = (lane & 16) ? 8 : 0;   // A layout: hi lanes hold K rel +8 / +24
#pragma unroll
  for (int nt = 0; nt < NTI_; ++nt) acc[nt] = (v8f)0.f;
#pragma unroll
  for (int kt = 0; kt < NKT_; ++kt) {
    const int kb = kbase + kt * 32 + koff;
    v8h alo = *(const v8h*)&act[row][kb];        // rel K {0..7}/{8..15}
    v8h ahi = *(const v8h*)&act[row][kb + 16];   // rel K {16..23}/{24..31}
    v16h a = __builtin_shufflevector(alo, ahi, 0, 1, 2, 3, 4, 5, 6, 7,
                                     8, 9, 10, 11, 12, 13, 14, 15);
#pragma unroll
    for (int nt = 0; nt < NTI_; ++nt) {
      v16h b = *(const v16h*)(wTiles + (size_t)(kt * NTI_ + nt) * 512 + lane * 16);
      acc[nt] = __builtin_amdgcn_wmma_f32_16x16x32_f16(
          false, a, false, b, (short)0, acc[nt], false, false);
    }
  }
}

// ------------------------------------------------------------------
// Fused main kernel: 256 threads = 8 waves, 128 points / block.
// Dynamic LDS: [0,94208) swizzled weights | act[128][200] f16 | disp[128][2] f32
// ------------------------------------------------------------------
__global__ __launch_bounds__(256, 2)
void fused_field(const float* __restrict__ inp,
                 const _Float16* __restrict__ ws,
                 WPtrs P,
                 float* __restrict__ out) {
  extern __shared__ __align__(16) char smem[];
  _Float16* wLds = (_Float16*)smem;
  _Float16 (*act)[200] = (_Float16 (*)[200])(smem + LDS_W_BYTES);
  float (*disp)[2] = (float (*)[2])(smem + LDS_W_BYTES + LDS_ACT_BYTES);

  const int tid   = threadIdx.x;
  const int lane  = tid & 31;
  const int wv    = tid >> 5;
  const int hiSel = (lane & 16) ? 1 : 0;
  const float invW = 2.0f / (AABB_MAX_F - AABB_MIN_F);
  const size_t gBase = (size_t)blockIdx.x * 128;

  // ---- stage swizzled weights global -> LDS (once per block) ----
  {
    const _Float16* gsrc = ws + WS_WEIGHTS_H;
#if USE_ASYNC_LDS
    // async copy (ASYNCcnt) overlaps with phase-1 sampling below
#pragma unroll 1
    for (int i = tid; i < W_CHUNKS; i += 256) {
      __builtin_amdgcn_global_load_async_to_lds_b128(
          (gas_v4i_p)(uintptr_t)(gsrc + (size_t)i * 8),
          (las_v4i_p)(unsigned)(uintptr_t)(wLds + (size_t)i * 8),
          0, 0);
    }
#else
#pragma unroll 1
    for (int i = tid; i < W_CHUNKS; i += 256) {
      *(v8h*)(wLds + (size_t)i * 8) = *(const v8h*)(gsrc + (size_t)i * 8);
    }
#endif
  }

  // bilinear sample 32 channels [c0,c0+32) of an HWC f16 plane -> act row (f16)
  auto sampleChans = [&](const _Float16* __restrict__ plane, float xn, float yn,
                         int c0, int row, int dstOff) {
    float gx = (xn + 1.f) * 0.5f * 511.f;
    float gy = (yn + 1.f) * 0.5f * 511.f;
    float x0f = floorf(gx), y0f = floorf(gy);
    int xi = (int)x0f, yi = (int)y0f;
    float wx1 = gx - x0f, wy1 = gy - y0f;
    float wx0 = 1.f - wx1, wy0 = 1.f - wy1;
    int   xs[4]  = {xi, xi + 1, xi, xi + 1};
    int   ys[4]  = {yi, yi, yi + 1, yi + 1};
    float wsc[4] = {wx0 * wy0, wx1 * wy0, wx0 * wy1, wx1 * wy1};
    float cw[4]; size_t co[4];
#pragma unroll
    for (int k = 0; k < 4; k++) {
      bool v = (xs[k] >= 0) && (xs[k] < RESO) && (ys[k] >= 0) && (ys[k] < RESO);
      int xc = xs[k] < 0 ? 0 : (xs[k] > RESO - 1 ? RESO - 1 : xs[k]);
      int yc = ys[k] < 0 ? 0 : (ys[k] > RESO - 1 ? RESO - 1 : ys[k]);
      cw[k] = v ? wsc[k] : 0.f;
      co[k] = ((size_t)yc * RESO + xc) * FEAT;
    }
#pragma unroll
    for (int ch = 0; ch < 32; ch += 8) {
      v8h p0 = *(const v8h*)(plane + co[0] + c0 + ch);
      v8h p1 = *(const v8h*)(plane + co[1] + c0 + ch);
      v8h p2 = *(const v8h*)(plane + co[2] + c0 + ch);
      v8h p3 = *(const v8h*)(plane + co[3] + c0 + ch);
      v8h r;
#pragma unroll
      for (int j = 0; j < 8; j++) {
        float a = cw[0] * (float)p0[j] + cw[1] * (float)p1[j] +
                  cw[2] * (float)p2[j] + cw[3] * (float)p3[j];
        r[j] = (_Float16)a;
      }
      *(v8h*)&act[row][dstOff + c0 + ch] = r;
    }
  };

  // ---------------- Phase 1: features + PE (2 threads per point) ----------
  {
    const int pl   = tid & 127;
    const int half = tid >> 7;   // channel half: 0 -> ch[0,32), 1 -> ch[32,64)
    const size_t gp = gBase + pl;
    float px = inp[gp * 3 + 0], py = inp[gp * 3 + 1], pt = inp[gp * 3 + 2];
    float xn = (px - AABB_MIN_F) * invW - 1.f;
    float yn = (py - AABB_MIN_F) * invW - 1.f;
    float tn = (pt - AABB_MIN_F) * invW - 1.f;
    sampleChans(ws + WS_DEFORM_H, xn, yn, half * 32, pl, 0);
    v8h z = {(_Float16)0.f, (_Float16)0.f, (_Float16)0.f, (_Float16)0.f,
             (_Float16)0.f, (_Float16)0.f, (_Float16)0.f, (_Float16)0.f};
    if (half == 0) {  // zero the K-pad [104,128) read by layer 0
      *(v8h*)&act[pl][104] = z;
      *(v8h*)&act[pl][112] = z;
      *(v8h*)&act[pl][120] = z;
    } else {          // zero the K-pad [168,200) read by skip layers; + PE
      *(v8h*)&act[pl][168] = z;
      *(v8h*)&act[pl][176] = z;
      *(v8h*)&act[pl][184] = z;
      *(v8h*)&act[pl][192] = z;
      const float w0 = 2.f * PI_F * tn;
      const float df = (1048576.f - 1.f) / 19.f;
#pragma unroll
      for (int i = 0; i < 20; i += 2) {
        float a0 = w0 * (1.f + (float)i * df);
        float a1 = w0 * (1.f + (float)(i + 1) * df);
        v2h s = {(_Float16)__sinf(a0), (_Float16)__sinf(a1)};
        v2h c = {(_Float16)__cosf(a0), (_Float16)__cosf(a1)};
        *(v2h*)&act[pl][64 + i] = s;
        *(v2h*)&act[pl][84 + i] = c;
      }
    }
  }

#if USE_ASYNC_LDS
  // wave-local: wait for our async weight-chunk copies to land in LDS
#if defined(__has_builtin) && __has_builtin(__builtin_amdgcn_s_wait_asynccnt)
  __builtin_amdgcn_s_wait_asynccnt(0);
#else
  asm volatile("s_wait_asynccnt 0x0" ::: "memory");
#endif
#endif
  __syncthreads();   // act + staged weights visible to all 8 waves

  // bias + relu + f16 store-back of a hidden layer (D layout: N=lane%16, M=r+8*hi)
  auto storeHidden = [&](const v8f* acc, int nti, const float* bias, int dstOff) {
#pragma unroll
    for (int nt = 0; nt < 4; ++nt) {
      if (nt >= nti) break;
      int ch = nt * 16 + (lane & 15);
      float bv = bias[ch];
#pragma unroll
      for (int r = 0; r < 8; ++r) {
        int prow = wv * 16 + r + hiSel * 8;
        float v = fmaxf(acc[nt][r] + bv, 0.f);
        act[prow][dstOff + ch] = (_Float16)v;
      }
    }
  };

  // ---------------- Phase 2: decoder MLP (wave-local, 16 pts / wave) ------
  v8f a4[4];
  v8f a1[1];
  wave_mm<4, 4>(act, wLds + 0,     0,   wv, lane, a4); storeHidden(a4, 4, P.B[0], 104);
  wave_mm<2, 4>(act, wLds + 8192,  104, wv, lane, a4); storeHidden(a4, 4, P.B[1], 104);
  wave_mm<6, 4>(act, wLds + 12288, 0,   wv, lane, a4); storeHidden(a4, 4, P.B[2], 104);
  wave_mm<2, 4>(act, wLds + 24576, 104, wv, lane, a4); storeHidden(a4, 4, P.B[3], 104);
  wave_mm<6, 4>(act, wLds + 28672, 0,   wv, lane, a4); storeHidden(a4, 4, P.B[4], 104);
  wave_mm<2, 1>(act, wLds + 40960, 104, wv, lane, a1);
  {
    int ch = lane & 15;
    if (ch < 2) {
      float bv = P.B[5][ch];
#pragma unroll
      for (int r = 0; r < 8; ++r) {
        int prow = wv * 16 + r + hiSel * 8;
        float v = a1[0][r] + bv;
        disp[prow][ch] = v;
        out[(gBase + prow) * 2 + ch] = v;   // output #1: displacement [N,2]
      }
    }
  }

  // ---------------- Phase 3: canonical sample (wave-local, in-order LDS) --
  {
    int pl = wv * 16 + (lane & 15);
    size_t gp = gBase + pl;
    float px = inp[gp * 3 + 0], py = inp[gp * 3 + 1];
    float xn = (px - AABB_MIN_F) * invW - 1.f;
    float yn = (py - AABB_MIN_F) * invW - 1.f;
    float cx = xn + disp[pl][0];
    float cy = yn + disp[pl][1];
    sampleChans(ws + WS_CANON_H, cx, cy, hiSel * 32, pl, 104);
  }

  // ---------------- Phase 4: canonical MLP --------------------------------
  wave_mm<2, 4>(act, wLds + 41984, 104, wv, lane, a4); storeHidden(a4, 4, P.B[6], 0);
  wave_mm<2, 1>(act, wLds + 46080, 0, wv, lane, a1);
  {
    int ch = lane & 15;
    if (ch < 3) {
      float bv = P.B[7][ch];
#pragma unroll
      for (int r = 0; r < 8; ++r) {
        int prow = wv * 16 + r + hiSel * 8;
        out[(size_t)NPTS * 2 + (gBase + prow) * 3 + ch] = a1[0][r] + bv;  // output #2: rgb [N,3]
      }
    }
  }
}

// ------------------------------------------------------------------
extern "C" void kernel_launch(void* const* d_in, const int* in_sizes, int n_in,
                              void* d_out, int out_size, void* d_ws, size_t ws_size,
                              hipStream_t stream) {
  (void)in_sizes; (void)out_size; (void)ws_size;
  const float* inp  = (const float*)d_in[0];
  const float* defp = (const float*)d_in[1];
  const float* canp = (const float*)d_in[2];

  WPtrs P;
  if (n_in >= 19) {
    // lists flattened to one input per array
    for (int i = 0; i < 6; i++) {
      P.W[i] = (const float*)d_in[3 + i];
      P.B[i] = (const float*)d_in[9 + i];
    }
    P.W[6] = (const float*)d_in[15];
    P.W[7] = (const float*)d_in[16];
    P.B[6] = (const float*)d_in[17];
    P.B[7] = (const float*)d_in[18];
  } else {
    // lists concatenated into single buffers
    const float* w  = (const float*)d_in[3];
    const float* b  = (const float*)d_in[4];
    const float* cw = (const float*)d_in[5];
    const float* cb = (const float*)d_in[6];
    const int wOff[6] = {0, 6656, 10752, 21504, 25600, 36352};
    for (int i = 0; i < 6; i++) { P.W[i] = w + wOff[i]; P.B[i] = b + i * 64; }
    P.W[6] = cw; P.W[7] = cw + 4096;
    P.B[6] = cb; P.B[7] = cb + 64;
  }

  _Float16* ws = (_Float16*)d_ws;  // needs ~67.3 MB: 2 f16 HWC planes + swizzled weights

  plane_to_hwc_f16<<<32768, 256, 0, stream>>>(defp, canp, ws);
  swizzle_weights<<<92, 512, 0, stream>>>(P, ws + WS_WEIGHTS_H);
  fused_field<<<NPTS / 128, 256, LDS_TOTAL, stream>>>(inp, ws, P, (float*)d_out);
}
